// SlideGraphArch_7610682048891
// MI455X (gfx1250) — matrix operand order repaired and checked by reading, not compiled, not run on hardware
//
#include <hip/hip_runtime.h>
#include <hip/hip_bf16.h>

typedef __attribute__((ext_vector_type(16))) _Float16 v16h;
typedef __attribute__((ext_vector_type(8)))  float    v8f;

#define GNN_N 100000
#define GNN_E 1600000
#define GNN_DF 256
#define GNN_H 64
#define GNN_T 4
#define GNN_G 8
#define BN_NB 512        // fixed partial-block count -> deterministic reduction
#define SC_BLOCKS 4096   // fixed block count for the edge scatter

// ---- order-preserving float<->uint encoding for atomic max ----
__device__ __forceinline__ unsigned enc_f32(float f) {
    unsigned u = __float_as_uint(f);
    return (u & 0x80000000u) ? ~u : (u | 0x80000000u);
}
__device__ __forceinline__ float dec_f32(unsigned u) {
    return (u & 0x80000000u) ? __uint_as_float(u & 0x7FFFFFFFu)
                             : __uint_as_float(~u);
}

// ---------------- init: zero agg + max codes ----------------
__global__ void k_init(float* __restrict__ agg, unsigned* __restrict__ maxb,
                       int n_agg, int n_max) {
    int id = blockIdx.x * blockDim.x + threadIdx.x;
    if (id < n_agg) agg[id] = 0.0f;
    if (id < n_max) maxb[id] = 0u;   // encodes "-inf-ish"
}

// ---------------- weight prep: W[K,64] f32 -> WH[64][K] f16 (transposed) ----------------
__global__ void k_prep_w(const float* __restrict__ W, _Float16* __restrict__ WH, int K) {
    int i = blockIdx.x * blockDim.x + threadIdx.x;
    if (i < 64 * K) {
        int n = i / K, k = i - n * K;
        WH[i] = (_Float16)W[(size_t)k * 64 + n];
    }
}

// ---- A fragment: 16 consecutive K per lane-half (matches B) ----
__device__ __forceinline__ v16h load_a_frag(const float* ap) {
    float4 a0 = *reinterpret_cast<const float4*>(ap);
    float4 a1 = *reinterpret_cast<const float4*>(ap + 4);
    float4 a2 = *reinterpret_cast<const float4*>(ap + 8);
    float4 a3 = *reinterpret_cast<const float4*>(ap + 12);
    v16h f;
    f[0]=(_Float16)a0.x; f[1]=(_Float16)a0.y; f[2]=(_Float16)a0.z; f[3]=(_Float16)a0.w;
    f[4]=(_Float16)a1.x; f[5]=(_Float16)a1.y; f[6]=(_Float16)a1.z; f[7]=(_Float16)a1.w;
    f[8]=(_Float16)a2.x; f[9]=(_Float16)a2.y; f[10]=(_Float16)a2.z; f[11]=(_Float16)a2.w;
    f[12]=(_Float16)a3.x; f[13]=(_Float16)a3.y; f[14]=(_Float16)a3.z; f[15]=(_Float16)a3.w;
    return f;
}

// ---------------- WMMA GEMM: C[M,64] = A[M,K] @ W[K,64] + bias ----------------
// 256 threads = 8 waves; each wave computes a 32x64 output strip (two 16-row
// tiles sharing B fragments); block = 256 rows. B (pre-transposed f16 [64][K])
// is staged to LDS with async global->LDS copies (ASYNCcnt path).
// __launch_bounds__(256,1): lift the occupancy-driven VGPR cap (8 f32 accums
// = 64 VGPRs + fragments) so the K-loop does not spill to scratch.
template <int K>
__global__ void __launch_bounds__(256, 1)
k_gemm_wmma(const float* __restrict__ A,
            const _Float16* __restrict__ WH,
            const float* __restrict__ bias,
            float* __restrict__ C,
            int Mrows) {
    __shared__ __align__(16) _Float16 Bh[64 * K];   // only LDS object -> offset 0
    const int tid = threadIdx.x;

    // ---- async stage Bh <= WH in 16B chunks (uniform trip count) ----
    constexpr int CHUNKS = 64 * K * 2 / 16;
    #pragma unroll
    for (int c = 0; c < CHUNKS / 256; ++c) {
        int i = c * 256 + tid;
        unsigned lds_off = (unsigned)i * 16u;
        unsigned long long ga =
            (unsigned long long)(const void*)((const char*)WH + (size_t)i * 16);
        asm volatile("global_load_async_to_lds_b128 %0, %1, off"
                     :: "v"(lds_off), "v"(ga) : "memory");
    }
    asm volatile("s_wait_asynccnt 0" ::: "memory");
    __syncthreads();

    const int wave = tid >> 5;
    const int lane = tid & 31;
    const int lrow = lane & 15;           // row / col within 16-tile
    const int hi   = (lane >> 4) & 1;     // upper half-wave flag

    const int base_row = blockIdx.x * 256 + wave * 32;
    int r0 = base_row + lrow;             // M-tile 0
    int r1 = base_row + 16 + lrow;        // M-tile 1
    if (r0 > Mrows - 1) r0 = Mrows - 1;   // clamp (stores are guarded)
    if (r1 > Mrows - 1) r1 = Mrows - 1;

    // accumulators: 2 M-tiles x 4 N-tiles; bias depends only on N
    v8f acc00, acc01, acc02, acc03, acc10, acc11, acc12, acc13;
    {
        float b0 = bias[0 + lrow], b1 = bias[16 + lrow];
        float b2 = bias[32 + lrow], b3 = bias[48 + lrow];
        #pragma unroll
        for (int r = 0; r < 8; ++r) {
            acc00[r] = b0; acc01[r] = b1; acc02[r] = b2; acc03[r] = b3;
            acc10[r] = b0; acc11[r] = b1; acc12[r] = b2; acc13[r] = b3;
        }
    }

    const int aoff = hi ? 16 : 0;   // A: lanes 16-31 take K+16
    const int boff = hi ? 16 : 0;   // B: lanes 16-31 take K+16
    const float*    arow0 = A + (size_t)r0 * K;
    const float*    arow1 = A + (size_t)r1 * K;
    const _Float16* bbase = &Bh[lrow * K + boff];

    #pragma unroll 2
    for (int kk = 0; kk < K; kk += 32) {
        v16h a0 = load_a_frag(arow0 + kk + aoff);
        v16h a1 = load_a_frag(arow1 + kk + aoff);

        const _Float16* bp = bbase + kk;
        v16h b0 = *reinterpret_cast<const v16h*>(bp + 0 * 16 * K);
        v16h b1 = *reinterpret_cast<const v16h*>(bp + 1 * 16 * K);
        v16h b2 = *reinterpret_cast<const v16h*>(bp + 2 * 16 * K);
        v16h b3 = *reinterpret_cast<const v16h*>(bp + 3 * 16 * K);

        acc00 = __builtin_amdgcn_wmma_f32_16x16x32_f16(false, a0, false, b0, (short)0, acc00, false, false);
        acc10 = __builtin_amdgcn_wmma_f32_16x16x32_f16(false, a1, false, b0, (short)0, acc10, false, false);
        acc01 = __builtin_amdgcn_wmma_f32_16x16x32_f16(false, a0, false, b1, (short)0, acc01, false, false);
        acc11 = __builtin_amdgcn_wmma_f32_16x16x32_f16(false, a1, false, b1, (short)0, acc11, false, false);
        acc02 = __builtin_amdgcn_wmma_f32_16x16x32_f16(false, a0, false, b2, (short)0, acc02, false, false);
        acc12 = __builtin_amdgcn_wmma_f32_16x16x32_f16(false, a1, false, b2, (short)0, acc12, false, false);
        acc03 = __builtin_amdgcn_wmma_f32_16x16x32_f16(false, a0, false, b3, (short)0, acc03, false, false);
        acc13 = __builtin_amdgcn_wmma_f32_16x16x32_f16(false, a1, false, b3, (short)0, acc13, false, false);
    }

    // C/D layout: VGPR r -> M = r (+8 for upper half-wave); N = lane&15
    const int rb0 = base_row + (hi ? 8 : 0);
    const int rb1 = rb0 + 16;
    #pragma unroll
    for (int r = 0; r < 8; ++r) {
        int row = rb0 + r;
        if (row < Mrows) {
            float* cr = C + (size_t)row * 64 + lrow;
            cr[0] = acc00[r]; cr[16] = acc01[r]; cr[32] = acc02[r]; cr[48] = acc03[r];
        }
    }
    #pragma unroll
    for (int r = 0; r < 8; ++r) {
        int row = rb1 + r;
        if (row < Mrows) {
            float* cr = C + (size_t)row * 64 + lrow;
            cr[0] = acc10[r]; cr[16] = acc11[r]; cr[32] = acc12[r]; cr[48] = acc13[r];
        }
    }
}

// ---------------- BN stats, stage 1: fixed-shape partials ----------------
__global__ void k_bn_part(const float* __restrict__ Z, float* __restrict__ part, int N) {
    const int col = threadIdx.x & 63;
    const int rg  = threadIdx.x >> 6;     // 0..3
    float s = 0.f, s2 = 0.f;
    for (int row = blockIdx.x * 4 + rg; row < N; row += BN_NB * 4) {
        float v = Z[(size_t)row * 64 + col];
        s += v; s2 += v * v;
    }
    __shared__ float sh[256], sh2[256];
    sh[threadIdx.x] = s; sh2[threadIdx.x] = s2;
    __syncthreads();
    if (rg == 0) {
        float a  = sh[col]  + sh[col + 64]  + sh[col + 128]  + sh[col + 192];
        float a2 = sh2[col] + sh2[col + 64] + sh2[col + 128] + sh2[col + 192];
        part[blockIdx.x * 128 + col]      = a;
        part[blockIdx.x * 128 + 64 + col] = a2;
    }
}

// ---------------- BN stats, stage 2: deterministic final sum ----------------
__global__ void k_bn_final(const float* __restrict__ part, float* __restrict__ stats) {
    int i = threadIdx.x;                  // 0..127: [0,64)=sum, [64,128)=sumsq
    float s = 0.f;
    for (int b = 0; b < BN_NB; ++b) s += part[b * 128 + i];
    stats[i] = s;
}

// ---------------- fused BN + ReLU + Linear(64->T) + node_pred + seg-max ----------------
__global__ void k_bn_np(float* __restrict__ Z,
                        const float* __restrict__ stats,
                        const float* __restrict__ gamma, const float* __restrict__ beta,
                        const float* __restrict__ Wl, const float* __restrict__ bl,
                        const int* __restrict__ batch,
                        float* __restrict__ node_out, unsigned* __restrict__ maxb,
                        int N, int accumulate) {
    __shared__ float s_mean[64], s_inv[64], s_g[64], s_b[64], s_W[256];
    int tid = threadIdx.x;
    if (tid < 64) {
        float m   = stats[tid] / (float)N;
        float var = stats[64 + tid] / (float)N - m * m;
        s_mean[tid] = m;
        s_inv[tid]  = rsqrtf(var + 1e-5f);
        s_g[tid] = gamma[tid];
        s_b[tid] = beta[tid];
    }
    s_W[tid] = Wl[tid];                   // blockDim == 256 == 64*4
    __syncthreads();

    int node = blockIdx.x * blockDim.x + tid;
    if (node >= N) return;

    float* zr = Z + (size_t)node * 64;
    float n0 = bl[0], n1 = bl[1], n2 = bl[2], n3 = bl[3];
    #pragma unroll 8
    for (int j = 0; j < 64; ++j) {
        float f = s_g[j] * (zr[j] - s_mean[j]) * s_inv[j] + s_b[j];
        f = fmaxf(f, 0.0f);
        zr[j] = f;                        // in-place: z -> relu(bn(z))
        n0 += f * s_W[j * 4 + 0];
        n1 += f * s_W[j * 4 + 1];
        n2 += f * s_W[j * 4 + 2];
        n3 += f * s_W[j * 4 + 3];
    }
    float* o = node_out + (size_t)node * 4;
    if (accumulate) { o[0] += n0; o[1] += n1; o[2] += n2; o[3] += n3; }
    else            { o[0]  = n0; o[1]  = n1; o[2]  = n2; o[3]  = n3; }

    int g4 = batch[node] * 4;
    atomicMax(&maxb[g4 + 0], enc_f32(n0));
    atomicMax(&maxb[g4 + 1], enc_f32(n1));
    atomicMax(&maxb[g4 + 2], enc_f32(n2));
    atomicMax(&maxb[g4 + 3], enc_f32(n3));
}

// ---------------- edge scatter: agg[dst] += f[src] ----------------
// One wave per edge iteration, grid-stride over edges with a global_prefetch_b8
// of the next iteration's source row to hide L2 gather latency.
__global__ void k_scatter(const float* __restrict__ F, const int* __restrict__ ei,
                          float* __restrict__ agg, int E, int nwaves) {
    int gid  = blockIdx.x * blockDim.x + threadIdx.x;
    int wid  = gid >> 5;
    int lane = gid & 31;
    for (int e = wid; e < E; e += nwaves) {
        int nx = e + nwaves;
        if (nx < E) {
            int ps = ei[nx];
            __builtin_prefetch(F + (size_t)ps * 64 + lane * 2, 0, 3); // global_prefetch_b8
        }
        int s = ei[e];
        int d = ei[E + e];
        const float2 v = *reinterpret_cast<const float2*>(F + (size_t)s * 64 + lane * 2);
        float* a = agg + (size_t)d * 64 + lane * 2;
        atomicAdd(a + 0, v.x);
        atomicAdd(a + 1, v.y);
    }
}

// ---------------- h = f + agg (into agg) ----------------
__global__ void k_add(float* __restrict__ agg, const float* __restrict__ F, int n) {
    int i = blockIdx.x * blockDim.x + threadIdx.x;
    if (i < n) agg[i] += F[i];
}

// ---------------- finalize: wsi = decode(max0) + decode(max1) ----------------
__global__ void k_finalize(const unsigned* __restrict__ maxb, float* __restrict__ wsi) {
    int i = threadIdx.x;                  // 0..31
    if (i < GNN_G * GNN_T)
        wsi[i] = dec_f32(maxb[i]) + dec_f32(maxb[GNN_G * GNN_T + i]);
}

extern "C" void kernel_launch(void* const* d_in, const int* in_sizes, int n_in,
                              void* d_out, int out_size, void* d_ws, size_t ws_size,
                              hipStream_t stream) {
    (void)in_sizes; (void)n_in; (void)out_size; (void)ws_size;

    const float* x       = (const float*)d_in[0];
    const int*   ei      = (const int*)  d_in[1];
    const int*   batch   = (const int*)  d_in[2];
    const float* W_first = (const float*)d_in[3];
    const float* b_first = (const float*)d_in[4];
    const float* bn1_g   = (const float*)d_in[5];
    const float* bn1_b   = (const float*)d_in[6];
    const float* W_lin0  = (const float*)d_in[7];
    const float* b_lin0  = (const float*)d_in[8];
    const float* W_conv  = (const float*)d_in[9];
    const float* b_conv  = (const float*)d_in[10];
    const float* bn2_g   = (const float*)d_in[11];
    const float* bn2_b   = (const float*)d_in[12];
    const float* W_lin1  = (const float*)d_in[13];
    const float* b_lin1  = (const float*)d_in[14];

    float*     F     = (float*)d_ws;                 // [N,64] z -> f -> z2 -> f2
    float*     AGG   = F   + (size_t)GNN_N * GNN_H;  // [N,64] agg -> h
    float*     PART  = AGG + (size_t)GNN_N * GNN_H;  // [BN_NB,128]
    float*     STATS = PART + (size_t)BN_NB * 128;   // [128]
    unsigned*  MAXB  = (unsigned*)(STATS + 128);     // [2,G,T]
    _Float16*  WH0   = (_Float16*)(MAXB + 2 * GNN_G * GNN_T); // [64,256] f16
    _Float16*  WH1   = WH0 + 64 * GNN_DF;                     // [64,64]  f16

    float* wsi_out  = (float*)d_out;                 // [8,4]
    float* node_out = (float*)d_out + GNN_G * GNN_T; // [N,4]

    const int NH = GNN_N * GNN_H;
    const int gemm_blocks = (GNN_N + 255) / 256;
    const int sc_waves = SC_BLOCKS * (256 / 32);

    // init + weight prep (f16, transposed)
    k_init  <<<(NH + 255) / 256, 256, 0, stream>>>(AGG, MAXB, NH, 2 * GNN_G * GNN_T);
    k_prep_w<<<(64 * GNN_DF + 255) / 256, 256, 0, stream>>>(W_first, WH0, GNN_DF);
    k_prep_w<<<(64 * GNN_H  + 255) / 256, 256, 0, stream>>>(W_conv,  WH1, GNN_H);

    // layer 0: z1 = x @ W_first + b  (WMMA, K=256)
    k_gemm_wmma<GNN_DF><<<gemm_blocks, 256, 0, stream>>>(x, WH0, b_first, F, GNN_N);
    k_bn_part <<<BN_NB, 256, 0, stream>>>(F, PART, GNN_N);
    k_bn_final<<<1, 128, 0, stream>>>(PART, STATS);
    k_bn_np   <<<(GNN_N + 255) / 256, 256, 0, stream>>>(
        F, STATS, bn1_g, bn1_b, W_lin0, b_lin0, batch,
        node_out, MAXB, GNN_N, /*accumulate=*/0);

    // GIN aggregation: agg[dst] += f[src];  h = f + agg
    k_scatter<<<SC_BLOCKS, 256, 0, stream>>>(F, ei, AGG, GNN_E, sc_waves);
    k_add    <<<(NH + 255) / 256, 256, 0, stream>>>(AGG, F, NH);

    // layer 1: z2 = h @ W_conv + b  (WMMA, K=64)
    k_gemm_wmma<GNN_H><<<gemm_blocks, 256, 0, stream>>>(AGG, WH1, b_conv, F, GNN_N);
    k_bn_part <<<BN_NB, 256, 0, stream>>>(F, PART, GNN_N);
    k_bn_final<<<1, 128, 0, stream>>>(PART, STATS);
    k_bn_np   <<<(GNN_N + 255) / 256, 256, 0, stream>>>(
        F, STATS, bn2_g, bn2_b, W_lin1, b_lin1, batch,
        node_out, MAXB + GNN_G * GNN_T, GNN_N, /*accumulate=*/1);

    // wsi_pred = segmax(np0) + segmax(np1)
    k_finalize<<<1, 32, 0, stream>>>(MAXB, wsi_out);
}
